// EdgeAwareGATFusion_64682207477952
// MI455X (gfx1250) — matrix operand order repaired and checked by reading, not compiled, not run on hardware
//
#include <hip/hip_runtime.h>
#include <hip/hip_bf16.h>
#include <math.h>

#define NA   8000
#define NLN  32000
#define NN   40000
#define DD   128
#define HH   8
#define DHD  16
#define E1   65536
#define EE   (3 * E1)
#define DFFN 256

typedef __attribute__((ext_vector_type(16))) __bf16 v16bfx;
typedef __attribute__((ext_vector_type(8)))  float  v8fx;

// ---------------------------------------------------------------------------
// WMMA tile GEMM, M=32: Out[32 x 16*ntiles] = A[32 x 32*kchunks] @ W^T (+bias)
// Two 16-row M-tiles share each B fragment (2 WMMAs per B load -> half the
// L2 weight traffic per FLOP vs M=16). A: LDS bf16 [32][lda]; W: global bf16
// row-major [Nout][K]. Layouts per CDNA5 ISA 7.12.2. OutT float or __bf16,
// optional fused ReLU in the epilogue.
// ---------------------------------------------------------------------------
template <typename OutT, bool RELU>
__device__ __forceinline__ void gemm_m32(
    const __bf16* A, int lda, int kchunks,
    const __bf16* __restrict__ W, int K, const float* __restrict__ bias,
    OutT* Out, int ldo, int ntiles, int lane)
{
  const int m16  = lane & 15;
  const int half = lane >> 4;
  for (int nt = 0; nt < ntiles; ++nt) {
    v8fx acc0 = {}, acc1 = {};
    const int n = nt * 16 + m16;
    for (int kc = 0; kc < kchunks; ++kc) {
      v16bfx a0, a1, b;
      const int kbb = kc * 32 + half * 8;      // B: lanes 0-15 K=0..7/16..23, 16-31 K=8..15/24..31
      #pragma unroll
      for (int j = 0; j < 8; ++j) {
        b[j]     = W[(size_t)n * K + kbb + j];
        b[8+j]   = W[(size_t)n * K + kbb + 16 + j];
      }
      const int kb = kc * 32 + half * 8;       // A: lanes 0-15 K=0..7/16..23, 16-31 K=8..15/24..31
      #pragma unroll
      for (int j = 0; j < 4; ++j) {
        a0[2*j]     = A[m16 * lda + kb + 2*j];
        a0[2*j+1]   = A[m16 * lda + kb + 2*j + 1];
        a0[8+2*j]   = A[m16 * lda + kb + 16 + 2*j];
        a0[8+2*j+1] = A[m16 * lda + kb + 16 + 2*j + 1];
        a1[2*j]     = A[(16 + m16) * lda + kb + 2*j];
        a1[2*j+1]   = A[(16 + m16) * lda + kb + 2*j + 1];
        a1[8+2*j]   = A[(16 + m16) * lda + kb + 16 + 2*j];
        a1[8+2*j+1] = A[(16 + m16) * lda + kb + 16 + 2*j + 1];
      }
      acc0 = __builtin_amdgcn_wmma_f32_16x16x32_bf16(
          false, a0, false, b, (short)0, acc0, false, false);
      acc1 = __builtin_amdgcn_wmma_f32_16x16x32_bf16(
          false, a1, false, b, (short)0, acc1, false, false);
    }
    const float bv = bias ? bias[n] : 0.0f;
    #pragma unroll
    for (int r = 0; r < 8; ++r) {              // C/D: VGPR r -> M = half*8+r, N = m16
      float v0 = acc0[r] + bv;
      float v1 = acc1[r] + bv;
      if (RELU) { v0 = fmaxf(v0, 0.f); v1 = fmaxf(v1, 0.f); }
      Out[(half * 8 + r) * ldo + n]        = (OutT)v0;
      Out[(16 + half * 8 + r) * ldo + n]   = (OutT)v1;
    }
  }
}

__device__ __forceinline__ void row_mean_inv(const float* p, int F, float& mu, float& inv)
{
  mu = 0.f;
  for (int f = 0; f < F; ++f) mu += p[f];
  mu *= (1.0f / F);
  float var = 0.f;
  for (int f = 0; f < F; ++f) { const float d = p[f] - mu; var += d * d; }
  inv = rsqrtf(var * (1.0f / F) + 1e-5f);
}

// ---------------------------------------------------------------------------
// Per-edge fused kernel, one wave per 32 edges:
// build_mem GEMM (K=384) -> LN/relu -> edge_update GEMM -> edge LN ->
// q/k/v GEMMs (bf16 epilogue) -> per-head logits -> atomic segment-softmax
// accumulation. Wo is hoisted past the scatter (linear op) to the node kernel.
// ---------------------------------------------------------------------------
__global__ void __launch_bounds__(32) gat_edge_kernel(
    const float* __restrict__ token,
    float* __restrict__ edge_attr,                      // [EE, DD] in/out
    const int* __restrict__ esrc, const int* __restrict__ edst,
    const __bf16* __restrict__ mpW, const float* __restrict__ mpb,
    const __bf16* __restrict__ euW, const float* __restrict__ eub,
    const __bf16* __restrict__ WqW, const __bf16* __restrict__ WkW,
    const __bf16* __restrict__ WvW,
    float* __restrict__ num, float* __restrict__ ssum)
{
  __shared__ __bf16 ldsA[32 * 384];     // [x_dst | x_src | e_attr] bf16
  __shared__ float  ldsF[32 * DD];      // f32 GEMM scratch
  __shared__ __bf16 ldsMB[32 * DD];     // mem (post LN/relu) bf16
  __shared__ __bf16 ldsQ[32 * DD];
  __shared__ __bf16 ldsK[32 * DD];
  __shared__ __bf16 ldsV[32 * DD];

  const int lane = threadIdx.x;
  const int e0   = blockIdx.x * 32;

  if (e0 + 32 < EE)                      // speculative prefetch of next tile's attrs
    __builtin_prefetch(&edge_attr[(size_t)(e0 + 32) * DD], 0, 1);

  // ---- gather: A = [x[dst] | x[src] | edge_attr] as bf16 ----
  for (int idx = lane; idx < 32 * DD; idx += 32) {
    const int r = idx >> 7, c = idx & (DD - 1);
    const int e = e0 + r;
    const int s = esrc[e], d = edst[e];
    ldsA[r * 384 + c]          = (__bf16)token[(size_t)d * DD + c];
    ldsA[r * 384 + DD + c]     = (__bf16)token[(size_t)s * DD + c];
    ldsA[r * 384 + 2 * DD + c] = (__bf16)edge_attr[(size_t)e * DD + c];
  }
  __syncthreads();

  // ---- mem = relu(ln(A @ mp_W^T + mp_b)) : 32x384x128, 192 WMMAs ----
  gemm_m32<float, false>(ldsA, 384, 12, mpW, 384, mpb, ldsF, DD, 8, lane);
  __syncthreads();
  {                                      // lane == row, all 32 lanes active
    float mu, inv; row_mean_inv(&ldsF[lane * DD], DD, mu, inv);
    for (int f = 0; f < DD; ++f)
      ldsMB[lane * DD + f] = (__bf16)fmaxf((ldsF[lane * DD + f] - mu) * inv, 0.f);
  }
  __syncthreads();

  // ---- delta = relu(ln(mem @ eu_W^T + eu_b)); edge_attr = ln(ea + delta) ----
  gemm_m32<float, false>(ldsMB, DD, 4, euW, DD, eub, ldsF, DD, 8, lane);
  __syncthreads();
  {
    float mu, inv; row_mean_inv(&ldsF[lane * DD], DD, mu, inv);
    const size_t eb = (size_t)(e0 + lane) * DD;
    for (int f = 0; f < DD; ++f)         // L2-hot re-read of this edge's attrs
      ldsF[lane * DD + f] =
          edge_attr[eb + f] + fmaxf((ldsF[lane * DD + f] - mu) * inv, 0.f);
    row_mean_inv(&ldsF[lane * DD], DD, mu, inv);
    for (int f = 0; f < DD; ++f)
      edge_attr[eb + f] = (ldsF[lane * DD + f] - mu) * inv;
  }
  __syncthreads();

  // ---- q = x[dst]@Wq^T, k = mem@Wk^T, v = mem@Wv^T (bf16 epilogues) ----
  gemm_m32<__bf16, false>(ldsA, 384, 4, WqW, DD, nullptr, ldsQ, DD, 8, lane);
  gemm_m32<__bf16, false>(ldsMB, DD, 4, WkW, DD, nullptr, ldsK, DD, 8, lane);
  gemm_m32<__bf16, false>(ldsMB, DD, 4, WvW, DD, nullptr, ldsV, DD, 8, lane);
  __syncthreads();

  // ---- per-head logits; accumulate exp-sum and exp-weighted v via atomics ----
  {
    const int r = lane;                  // one edge per lane
    const int e = e0 + r;
    const int dn = edst[e];
    for (int h = 0; h < HH; ++h) {
      float lg = 0.f;
      for (int d = 0; d < DHD; ++d)
        lg += (float)ldsQ[r * DD + h * DHD + d] * (float)ldsK[r * DD + h * DHD + d];
      const float ex = __expf(lg * 0.25f);        // 1/sqrt(DH) = 0.25
      atomicAdd(&ssum[dn * HH + h], ex);
      for (int d = 0; d < DHD; ++d)
        atomicAdd(&num[(size_t)dn * DD + h * DHD + d],
                  ex * (float)ldsV[r * DD + h * DHD + d]);
    }
  }
}

// ---------------------------------------------------------------------------
// Per-node fused kernel, one wave per 32 nodes:
// aggr = (num/ssum) @ Wo^T; x = ln(x+aggr); FFN (fused relu->bf16); ln.
// ---------------------------------------------------------------------------
__global__ void __launch_bounds__(32) gat_node_kernel(
    float* __restrict__ token,                           // [NN, DD] in/out
    const float* __restrict__ num, const float* __restrict__ ssum,
    const __bf16* __restrict__ WoW,
    const __bf16* __restrict__ f1W, const float* __restrict__ f1b,
    const __bf16* __restrict__ f2W, const float* __restrict__ f2b)
{
  __shared__ __bf16 ldsAB[32 * DD];
  __shared__ float  ldsX[32 * DD];
  __shared__ __bf16 ldsXB[32 * DD];
  __shared__ __bf16 ldsHB[32 * DFFN];
  __shared__ float  ldsT[32 * DD];

  const int lane = threadIdx.x;
  const int n0   = blockIdx.x * 32;

  for (int idx = lane; idx < 32 * DD; idx += 32) {
    const int r = idx >> 7, c = idx & (DD - 1);
    const int n = n0 + r;
    const int h = c >> 4;
    ldsAB[r * DD + c] =
        (__bf16)(num[(size_t)n * DD + c] / (ssum[n * HH + h] + 1e-16f));
  }
  __syncthreads();
  gemm_m32<float, false>(ldsAB, DD, 4, WoW, DD, nullptr, ldsT, DD, 8, lane);
  __syncthreads();
  {                                      // lane == row
    const size_t nb = (size_t)(n0 + lane) * DD;
    for (int f = 0; f < DD; ++f)
      ldsX[lane * DD + f] = token[nb + f] + ldsT[lane * DD + f];
    float mu, inv; row_mean_inv(&ldsX[lane * DD], DD, mu, inv);
    for (int f = 0; f < DD; ++f) {
      const float v = (ldsX[lane * DD + f] - mu) * inv;
      ldsX[lane * DD + f]  = v;
      ldsXB[lane * DD + f] = (__bf16)v;
    }
  }
  __syncthreads();
  gemm_m32<__bf16, true>(ldsXB, DD, 4, f1W, DD, f1b, ldsHB, DFFN, 16, lane);
  __syncthreads();
  gemm_m32<float, false>(ldsHB, DFFN, 8, f2W, DFFN, f2b, ldsT, DD, 8, lane);
  __syncthreads();
  {
    for (int f = 0; f < DD; ++f)
      ldsX[lane * DD + f] += ldsT[lane * DD + f];
    float mu, inv; row_mean_inv(&ldsX[lane * DD], DD, mu, inv);
    const size_t nb = (size_t)(n0 + lane) * DD;
    for (int f = 0; f < DD; ++f)
      token[nb + f] = (ldsX[lane * DD + f] - mu) * inv;
  }
}

// ---------------------------------------------------------------------------
// DDC refinement: a = softmax(gelu(rms(x)@W1+b1)@W2+b2); token = sum a_k*prev_k
// ---------------------------------------------------------------------------
__global__ void __launch_bounds__(128) ddc_kernel(
    float* __restrict__ token,
    const float* __restrict__ prev0, const float* __restrict__ prev1, int k,
    const float* __restrict__ W1, const float* __restrict__ b1,
    const float* __restrict__ W2, const float* __restrict__ b2)
{
  __shared__ float xh[DD];
  __shared__ float av[DD];
  __shared__ float red[DD];
  __shared__ float wts[4];
  const int t = threadIdx.x;
  const size_t nb = (size_t)blockIdx.x * DD;
  const float xv = token[nb + t];
  red[t] = xv * xv;
  __syncthreads();
  for (int s = 64; s > 0; s >>= 1) {
    if (t < s) red[t] += red[t + s];
    __syncthreads();
  }
  const float rms = sqrtf(red[0]) * 0.08838834764f;   // /sqrt(128)
  xh[t] = xv / (rms + 1e-8f);
  __syncthreads();
  float acc = b1[t];
  for (int j = 0; j < DD; ++j) acc += xh[j] * W1[t * DD + j];
  av[t] = 0.5f * acc * (1.f + erff(acc * 0.70710678f));
  __syncthreads();
  if (t < k) {
    float l = b2[t];
    for (int j = 0; j < DD; ++j) l += av[j] * W2[t * DD + j];
    wts[t] = l;
  }
  __syncthreads();
  if (t == 0) {
    float m = wts[0];
    for (int i = 1; i < k; ++i) m = fmaxf(m, wts[i]);
    float s = 0.f;
    for (int i = 0; i < k; ++i) { wts[i] = __expf(wts[i] - m); s += wts[i]; }
    for (int i = 0; i < k; ++i) wts[i] /= s;
  }
  __syncthreads();
  float o = wts[0] * prev0[nb + t];
  if (k > 1) o += wts[1] * prev1[nb + t];
  token[nb + t] = o;
}

// ---- prep kernels -------------------------------------------------------
__global__ void __launch_bounds__(256) cvt_bf16_kernel(
    const float* __restrict__ in, __bf16* __restrict__ out, int n)
{
  const int i = blockIdx.x * 256 + threadIdx.x;
  if (i < n) out[i] = (__bf16)in[i];
}

__global__ void __launch_bounds__(256) concat_edges_kernel(
    const int* __restrict__ a2a, const int* __restrict__ l2l,
    const int* __restrict__ a2l, int* __restrict__ esrc, int* __restrict__ edst)
{
  const int e = blockIdx.x * 256 + threadIdx.x;
  if (e >= EE) return;
  const int* p = (e < E1) ? a2a : ((e < 2 * E1) ? l2l : a2l);
  const int o  = (e < E1) ? e   : ((e < 2 * E1) ? e - E1 : e - 2 * E1);
  esrc[e] = p[o];
  edst[e] = p[E1 + o];
}

__global__ void __launch_bounds__(128) a2l_rpe_kernel(
    const float* __restrict__ rpe, const float* __restrict__ W,
    const float* __restrict__ b, float* __restrict__ out)
{
  const int e = blockIdx.x * 128 + threadIdx.x;
  if (e >= E1) return;
  float r[5];
  #pragma unroll
  for (int j = 0; j < 5; ++j) r[j] = rpe[e * 5 + j];
  float y[DD];
  float mu = 0.f;
  for (int d = 0; d < DD; ++d) {
    float t = b[d];
    #pragma unroll
    for (int j = 0; j < 5; ++j) t += r[j] * W[d * 5 + j];
    y[d] = t; mu += t;
  }
  mu *= (1.f / DD);
  float var = 0.f;
  for (int d = 0; d < DD; ++d) { const float dv = y[d] - mu; var += dv * dv; }
  const float inv = rsqrtf(var * (1.f / DD) + 1e-5f);
  const size_t ob = (size_t)e * DD;
  for (int d = 0; d < DD; ++d)
    out[ob + d] = fmaxf((y[d] - mu) * inv, 0.f);
}

// ---------------------------------------------------------------------------
extern "C" void kernel_launch(void* const* d_in, const int* in_sizes, int n_in,
                              void* d_out, int out_size, void* d_ws, size_t ws_size,
                              hipStream_t stream)
{
  (void)in_sizes; (void)n_in; (void)out_size; (void)ws_size;
  const float* actors   = (const float*)d_in[0];
  const float* lanesF   = (const float*)d_in[1];
  const float* a2a_attr = (const float*)d_in[2];
  const float* l2l_attr = (const float*)d_in[3];
  const float* a2l_rpe  = (const float*)d_in[4];
  const float* rpe_W    = (const float*)d_in[5];
  const float* rpe_b    = (const float*)d_in[6];
  const float* mp_W     = (const float*)d_in[7];
  const float* mp_b     = (const float*)d_in[8];
  const float* eu_W     = (const float*)d_in[9];
  const float* eu_b     = (const float*)d_in[10];
  const float* WqF      = (const float*)d_in[11];
  const float* WkF      = (const float*)d_in[12];
  const float* WvF      = (const float*)d_in[13];
  const float* WoF      = (const float*)d_in[14];
  const float* f1WF     = (const float*)d_in[15];
  const float* f1b      = (const float*)d_in[16];
  const float* f2WF     = (const float*)d_in[17];
  const float* f2b      = (const float*)d_in[18];
  const float* rf_W1[2] = {(const float*)d_in[19], (const float*)d_in[23]};
  const float* rf_b1[2] = {(const float*)d_in[20], (const float*)d_in[24]};
  const float* rf_W2[2] = {(const float*)d_in[21], (const float*)d_in[25]};
  const float* rf_b2[2] = {(const float*)d_in[22], (const float*)d_in[26]};
  const int* a2a_e = (const int*)d_in[27];
  const int* l2l_e = (const int*)d_in[28];
  const int* a2l_e = (const int*)d_in[29];

  // workspace carving (256B aligned)
  char* ws = (char*)d_ws;
  size_t off = 0;
  auto take = [&](size_t bytes) -> char* {
    char* p = ws + off;
    off = (off + bytes + 255) & ~(size_t)255;
    return p;
  };
  float* token = (float*)take((size_t)NN * DD * 4);
  float* tok0  = (float*)take((size_t)NN * DD * 4);
  float* tok1  = (float*)take((size_t)NN * DD * 4);
  float* eattr = (float*)take((size_t)EE * DD * 4);
  float* num   = (float*)take((size_t)NN * DD * 4);
  float* ssum  = (float*)take((size_t)NN * HH * 4);
  int*   esrc  = (int*)take((size_t)EE * 4);
  int*   edst  = (int*)take((size_t)EE * 4);
  __bf16* mpWb = (__bf16*)take((size_t)3 * DD * 3 * DD * 2);
  __bf16* euWb = (__bf16*)take((size_t)3 * DD * DD * 2);
  __bf16* WqWb = (__bf16*)take((size_t)3 * DD * DD * 2);
  __bf16* WkWb = (__bf16*)take((size_t)3 * DD * DD * 2);
  __bf16* WvWb = (__bf16*)take((size_t)3 * DD * DD * 2);
  __bf16* WoWb = (__bf16*)take((size_t)3 * DD * DD * 2);
  __bf16* f1Wb = (__bf16*)take((size_t)3 * DFFN * DD * 2);
  __bf16* f2Wb = (__bf16*)take((size_t)3 * DD * DFFN * 2);

  // ---- prep: token / edge_attr assembly, edge concat, bf16 weights ----
  hipMemcpyAsync(token, actors, (size_t)NA * DD * 4, hipMemcpyDeviceToDevice, stream);
  hipMemcpyAsync(token + (size_t)NA * DD, lanesF, (size_t)NLN * DD * 4,
                 hipMemcpyDeviceToDevice, stream);
  hipMemcpyAsync(eattr, a2a_attr, (size_t)E1 * DD * 4, hipMemcpyDeviceToDevice, stream);
  hipMemcpyAsync(eattr + (size_t)E1 * DD, l2l_attr, (size_t)E1 * DD * 4,
                 hipMemcpyDeviceToDevice, stream);
  a2l_rpe_kernel<<<E1 / 128, 128, 0, stream>>>(a2l_rpe, rpe_W, rpe_b,
                                               eattr + (size_t)2 * E1 * DD);
  concat_edges_kernel<<<(EE + 255) / 256, 256, 0, stream>>>(a2a_e, l2l_e, a2l_e,
                                                            esrc, edst);
  {
    const int n1 = 3 * DD * 3 * DD, n2 = 3 * DD * DD, n3 = 3 * DFFN * DD;
    cvt_bf16_kernel<<<(n1 + 255) / 256, 256, 0, stream>>>(mp_W, mpWb, n1);
    cvt_bf16_kernel<<<(n2 + 255) / 256, 256, 0, stream>>>(eu_W, euWb, n2);
    cvt_bf16_kernel<<<(n2 + 255) / 256, 256, 0, stream>>>(WqF, WqWb, n2);
    cvt_bf16_kernel<<<(n2 + 255) / 256, 256, 0, stream>>>(WkF, WkWb, n2);
    cvt_bf16_kernel<<<(n2 + 255) / 256, 256, 0, stream>>>(WvF, WvWb, n2);
    cvt_bf16_kernel<<<(n2 + 255) / 256, 256, 0, stream>>>(WoF, WoWb, n2);
    cvt_bf16_kernel<<<(n3 + 255) / 256, 256, 0, stream>>>(f1WF, f1Wb, n3);
    cvt_bf16_kernel<<<(n3 + 255) / 256, 256, 0, stream>>>(f2WF, f2Wb, n3);
  }
  hipMemcpyAsync(tok0, token, (size_t)NN * DD * 4, hipMemcpyDeviceToDevice, stream);

  // ---- layers ----
  for (int i = 0; i < 3; ++i) {
    hipMemsetAsync(num, 0, (size_t)NN * DD * 4, stream);
    hipMemsetAsync(ssum, 0, (size_t)NN * HH * 4, stream);
    gat_edge_kernel<<<EE / 32, 32, 0, stream>>>(
        token, eattr, esrc, edst,
        mpWb + (size_t)i * DD * 3 * DD, mp_b + (size_t)i * DD,
        euWb + (size_t)i * DD * DD,     eu_b + (size_t)i * DD,
        WqWb + (size_t)i * DD * DD, WkWb + (size_t)i * DD * DD,
        WvWb + (size_t)i * DD * DD,
        num, ssum);
    gat_node_kernel<<<NN / 32, 32, 0, stream>>>(
        token, num, ssum,
        WoWb + (size_t)i * DD * DD,
        f1Wb + (size_t)i * DFFN * DD, f1b + (size_t)i * DFFN,
        f2Wb + (size_t)i * DD * DFFN, f2b + (size_t)i * DD);
    if (i == 0) {
      ddc_kernel<<<NN, 128, 0, stream>>>(token, tok0, tok0, 1,
                                         rf_W1[0], rf_b1[0], rf_W2[0], rf_b2[0]);
      hipMemcpyAsync(tok1, token, (size_t)NN * DD * 4, hipMemcpyDeviceToDevice, stream);
    } else if (i == 1) {
      ddc_kernel<<<NN, 128, 0, stream>>>(token, tok0, tok1, 2,
                                         rf_W1[1], rf_b1[1], rf_W2[1], rf_b2[1]);
    }
  }

  hipMemcpyAsync(d_out, token, (size_t)NN * DD * 4, hipMemcpyDeviceToDevice, stream);
}
